// FirefoxIssueGraphSAGE_91268055040046
// MI455X (gfx1250) — compile-verified, matrix operand order checked by reading, not run on hardware
//
#include <hip/hip_runtime.h>
#include <hip/hip_bf16.h>
#include <math.h>

#define N_NODES 100000
#define N_EDGES 625000
#define D_IN    128
#define D_K     256          // concat K dim: [mean | x]
#define D_HID   256
#define N_TILES (N_NODES / 16)   // 6250, exact

typedef __bf16 v16bf __attribute__((ext_vector_type(16)));
typedef __bf16 v8bf  __attribute__((ext_vector_type(8)));
typedef float  v8f   __attribute__((ext_vector_type(8)));

// ---------------------------------------------------------------------------
// Kernel 1: edge scatter. One wave per edge; lane handles 4 features.
// agg[dst] += x[src]; cnt[dst] += 1
// ---------------------------------------------------------------------------
__global__ __launch_bounds__(256)
void sage_edge_scatter(const float* __restrict__ x,
                       const long long* __restrict__ ei,
                       float* __restrict__ agg, float* __restrict__ cnt)
{
    unsigned gtid = blockIdx.x * blockDim.x + threadIdx.x;
    unsigned wid  = gtid >> 5;
    unsigned lane = gtid & 31;
    if (wid >= N_EDGES) return;
    long long s = ei[wid];
    long long d = ei[N_EDGES + wid];
    const float4 v = *(const float4*)(x + (size_t)s * D_IN + lane * 4);
    float* a = agg + (size_t)d * D_IN + lane * 4;
    atomicAdd(a + 0, v.x);
    atomicAdd(a + 1, v.y);
    atomicAdd(a + 2, v.z);
    atomicAdd(a + 3, v.w);
    if (lane == 0) atomicAdd(cnt + d, 1.0f);
}

// ---------------------------------------------------------------------------
// Kernel 2: mean = agg / max(cnt,1); pack A = [mean | x] as bf16, row-major
// [N_NODES][256]. One wave per node; lane handles 4 features.
// ---------------------------------------------------------------------------
__global__ __launch_bounds__(256)
void sage_build_a(const float* __restrict__ x,
                  const float* __restrict__ agg, const float* __restrict__ cnt,
                  __hip_bfloat16* __restrict__ Abf)
{
    unsigned gtid = blockIdx.x * blockDim.x + threadIdx.x;
    unsigned node = gtid >> 5;
    unsigned lane = gtid & 31;
    if (node >= N_NODES) return;
    float c = cnt[node];
    c = c > 1.0f ? c : 1.0f;
    float inv = 1.0f / c;
    const float4 av = *(const float4*)(agg + (size_t)node * D_IN + lane * 4);
    const float4 xv = *(const float4*)(x   + (size_t)node * D_IN + lane * 4);
    __hip_bfloat16* row = Abf + (size_t)node * D_K;
    row[lane * 4 + 0] = __float2bfloat16(av.x * inv);
    row[lane * 4 + 1] = __float2bfloat16(av.y * inv);
    row[lane * 4 + 2] = __float2bfloat16(av.z * inv);
    row[lane * 4 + 3] = __float2bfloat16(av.w * inv);
    row[D_IN + lane * 4 + 0] = __float2bfloat16(xv.x);
    row[D_IN + lane * 4 + 1] = __float2bfloat16(xv.y);
    row[D_IN + lane * 4 + 2] = __float2bfloat16(xv.z);
    row[D_IN + lane * 4 + 3] = __float2bfloat16(xv.w);
}

// ---------------------------------------------------------------------------
// Kernel 3: pack weights N-major (Wt[n][k] = k<128 ? w_l[k][n] : w_r[k-128][n])
// so B fragments are contiguous 16B loads.
// ---------------------------------------------------------------------------
__global__ __launch_bounds__(256)
void sage_build_w(const float* __restrict__ w_l, const float* __restrict__ w_r,
                  __hip_bfloat16* __restrict__ Wt)
{
    unsigned t = blockIdx.x * blockDim.x + threadIdx.x;  // 65536
    unsigned n = t >> 8;
    unsigned k = t & 255;
    float v = (k < 128) ? w_l[k * D_HID + n] : w_r[(k - 128) * D_HID + n];
    Wt[n * 256 + k] = __float2bfloat16(v);
}

// ---------------------------------------------------------------------------
// Fragment loader: 16-bit WMMA layout (ISA 7.12.2).
// lane<16 holds idx=lane, K = {kk..kk+7, kk+16..kk+23}
// lane>=16 holds idx=lane-16, K = {kk+8..kk+15, kk+24..kk+31}
// base points at start of the (row/col) vector of 256 bf16 (LDS or global).
// ---------------------------------------------------------------------------
__device__ __forceinline__ v16bf load_frag16(const __bf16* base, int kk, int half)
{
    v8bf lo = *(const v8bf*)(base + kk + half * 8);
    v8bf hi = *(const v8bf*)(base + kk + 16 + half * 8);
    return __builtin_shufflevector(lo, hi, 0,1,2,3,4,5,6,7,8,9,10,11,12,13,14,15);
}

// ---------------------------------------------------------------------------
// Kernel 4: fused GEMM (WMMA bf16) + bias + ReLU + WMMA heads + log_softmax.
// 256 threads = 8 waves. Weight B-fragments live in registers (128 VGPRs/wave,
// loaded once from global). Heads done with split-K WMMA across the 8 waves.
// Grid-stride over 6250 node tiles of 16 rows.
// ---------------------------------------------------------------------------
__global__ __launch_bounds__(256)
void sage_gemm_fused(const __bf16* __restrict__ Abf,
                     const __bf16* __restrict__ Wt,
                     const float* __restrict__ b_l,
                     const float* __restrict__ wp, const float* __restrict__ bp,
                     const float* __restrict__ ws, const float* __restrict__ bs,
                     float* __restrict__ out)
{
    __shared__ __bf16 Alds[16 * 256];       // 8 KB: A tile (bf16)
    __shared__ __bf16 hbf [16 * 256];       // 8 KB: hidden (bf16, post-ReLU)
    __shared__ __bf16 W13t[16 * 256];       // 8 KB: [wp|ws|0] N-major bf16
    __shared__ float  plds[8 * 16 * 16];    // 8 KB: per-wave head partials
    __shared__ float  lglds[16 * 16];       // 1 KB: logits (13 used/row)

    const int tid  = threadIdx.x;
    const int lane = tid & 31;
    const int wave = tid >> 5;
    const int half = lane >> 4;
    const int m    = lane & 15;

    // --- build head-weight tile in LDS: W13t[j][k], j 0..6 = wp, 7..12 = ws ---
    for (int i = tid; i < 16 * 256; i += 256) {
        const int j = i >> 8;
        const int k = i & 255;
        float v = (j < 7) ? wp[k * 7 + j] : (j < 13 ? ws[k * 6 + (j - 7)] : 0.0f);
        W13t[i] = (__bf16)v;
    }

    // --- hoist main-GEMM B fragments into registers (once per block) ---
    // wave owns column tiles n0 = wave*32 and wave*32+16
    v16bf bfrag[2][8];
    #pragma unroll
    for (int nt = 0; nt < 2; ++nt) {
        const __bf16* brow = Wt + (size_t)(wave * 32 + nt * 16 + m) * 256;
        #pragma unroll
        for (int k = 0; k < 8; ++k)
            bfrag[nt][k] = load_frag16(brow, k * 32, half);
    }

    // per-lane biases in registers
    const float bias0 = b_l[wave * 32 + m];
    const float bias1 = b_l[wave * 32 + 16 + m];
    const int   hrow  = tid >> 4;           // reduction indices
    const int   hcol  = tid & 15;
    const float hbias = (hcol < 7) ? bp[hcol] : (hcol < 13 ? bs[hcol - 7] : 0.0f);

    __syncthreads();

    for (int tile = blockIdx.x; tile < N_TILES; tile += gridDim.x) {
        // prefetch next tile's A rows into cache (global_prefetch_b8)
        {
            const int nxt = tile + gridDim.x;
            if (nxt < N_TILES)
                __builtin_prefetch(Abf + (size_t)nxt * 16 * D_K + tid * 32, 0, 3);
        }
        // --- load A tile (16 x 256 bf16 = 8 KB) ---
        {
            const uint4* src = (const uint4*)(Abf + (size_t)tile * 16 * D_K);
            uint4* dst = (uint4*)Alds;
            for (int i = tid; i < 16 * 256 * 2 / 16; i += 256) dst[i] = src[i];
        }
        __syncthreads();

        // --- main GEMM: shared A fragment feeds both column-tile accs ---
        {
            v8f acc0 = {}, acc1 = {};
            const __bf16* arow = Alds + m * 256;
            #pragma unroll
            for (int k = 0; k < 8; ++k) {
                v16bf a = load_frag16(arow, k * 32, half);
                acc0 = __builtin_amdgcn_wmma_f32_16x16x32_bf16(
                        false, a, false, bfrag[0][k], (short)0, acc0, false, false);
                acc1 = __builtin_amdgcn_wmma_f32_16x16x32_bf16(
                        false, a, false, bfrag[1][k], (short)0, acc1, false, false);
            }
            const int col0 = wave * 32 + m;
            #pragma unroll
            for (int r = 0; r < 8; ++r) {            // C: row = r + 8*half
                float v0 = acc0[r] + bias0;
                float v1 = acc1[r] + bias1;
                hbf[(r + half * 8) * 256 + col0]      = (__bf16)(v0 > 0.0f ? v0 : 0.0f);
                hbf[(r + half * 8) * 256 + col0 + 16] = (__bf16)(v1 > 0.0f ? v1 : 0.0f);
            }
        }
        __syncthreads();

        // --- heads via split-K WMMA: wave w handles K chunk w*32 ---
        {
            v16bf a = load_frag16(hbf  + m * 256, wave * 32, half);
            v16bf b = load_frag16(W13t + m * 256, wave * 32, half);
            v8f acc = {};
            acc = __builtin_amdgcn_wmma_f32_16x16x32_bf16(
                    false, a, false, b, (short)0, acc, false, false);
            #pragma unroll
            for (int r = 0; r < 8; ++r)
                plds[wave * 256 + (r + half * 8) * 16 + m] = acc[r];
        }
        __syncthreads();

        // --- reduce 8 partials + bias -> logits ---
        if (hcol < 13) {
            float s = hbias;
            #pragma unroll
            for (int w = 0; w < 8; ++w) s += plds[w * 256 + tid];
            lglds[tid] = s;
        }
        __syncthreads();

        // --- log_softmax + store: t<16 -> pr row t, t in [16,32) -> sv row ---
        if (tid < 32) {
            const int  row   = tid & 15;
            const bool is_pr = tid < 16;
            const int  nc    = is_pr ? 7 : 6;
            const float* base = lglds + row * 16 + (is_pr ? 0 : 7);
            float mx = base[0];
            for (int j = 1; j < nc; ++j) mx = fmaxf(mx, base[j]);
            float sum = 0.0f;
            for (int j = 0; j < nc; ++j) sum += expf(base[j] - mx);
            const float lse = logf(sum);
            const size_t node = (size_t)tile * 16 + row;
            float* o = is_pr ? (out + node * 7)
                             : (out + (size_t)N_NODES * 7 + node * 6);
            for (int j = 0; j < nc; ++j) o[j] = base[j] - mx - lse;
        }
        __syncthreads();   // protect Alds/hbf/plds/lglds for next iteration
    }
}

// ---------------------------------------------------------------------------
// Launch
// ---------------------------------------------------------------------------
extern "C" void kernel_launch(void* const* d_in, const int* in_sizes, int n_in,
                              void* d_out, int out_size, void* d_ws, size_t ws_size,
                              hipStream_t stream) {
    const float*     x   = (const float*)d_in[0];
    const long long* ei  = (const long long*)d_in[1];
    const float*     w_l = (const float*)d_in[2];
    const float*     b_l = (const float*)d_in[3];
    const float*     w_r = (const float*)d_in[4];
    const float*     w_p = (const float*)d_in[5];
    const float*     b_p = (const float*)d_in[6];
    const float*     w_s = (const float*)d_in[7];
    const float*     b_s = (const float*)d_in[8];
    float* out = (float*)d_out;

    // workspace layout
    float* agg = (float*)d_ws;                               // 100000*128 f32
    float* cnt = agg + (size_t)N_NODES * D_IN;               // 100000 f32
    __hip_bfloat16* Abf = (__hip_bfloat16*)(cnt + N_NODES);  // 100000*256 bf16
    __hip_bfloat16* Wt  = Abf + (size_t)N_NODES * D_K;       // 256*256 bf16

    // zero accumulators (agg + cnt in one contiguous memset)
    hipMemsetAsync(agg, 0, ((size_t)N_NODES * D_IN + N_NODES) * sizeof(float), stream);

    // 1) edge scatter: one wave per edge
    {
        unsigned threads = N_EDGES * 32u;
        sage_edge_scatter<<<(threads + 255) / 256, 256, 0, stream>>>(x, ei, agg, cnt);
    }
    // 2) mean + bf16 pack of [mean|x]
    {
        unsigned threads = N_NODES * 32u;
        sage_build_a<<<(threads + 255) / 256, 256, 0, stream>>>(x, agg, cnt, Abf);
    }
    // 3) weight pack (N-major bf16)
    sage_build_w<<<(256 * 256) / 256, 256, 0, stream>>>(w_l, w_r, Wt);

    // 4) fused WMMA GEMM + WMMA heads + log_softmax
    sage_gemm_fused<<<640, 256, 0, stream>>>((const __bf16*)Abf, (const __bf16*)Wt,
                                             b_l, w_p, b_p, w_s, b_s, out);
}
